// Attention_embedding_layer_4312147165640
// MI455X (gfx1250) — compile-verified, matrix operand order checked by reading, not run on hardware
//
#include <hip/hip_runtime.h>
#include <math.h>

#define B_TOT 2048
#define FDIM  256
#define MIDD  128
#define EDIM  8
#define HH    2
#define HDD   4

typedef __attribute__((ext_vector_type(2)))  float    v2f;
typedef __attribute__((ext_vector_type(8)))  float    v8f;
typedef __attribute__((ext_vector_type(16))) _Float16 v16h;

__device__ __forceinline__ float grp16_max(float v) {
  v = fmaxf(v, __shfl_xor(v, 8, 32));
  v = fmaxf(v, __shfl_xor(v, 4, 32));
  v = fmaxf(v, __shfl_xor(v, 2, 32));
  v = fmaxf(v, __shfl_xor(v, 1, 32));
  return v;
}
__device__ __forceinline__ float grp16_sum(float v) {
  v += __shfl_xor(v, 8, 32);
  v += __shfl_xor(v, 4, 32);
  v += __shfl_xor(v, 2, 32);
  v += __shfl_xor(v, 1, 32);
  return v;
}

// ---------------------------------------------------------------------------
// Kernel 1: one workgroup (256 thr = 8 waves) per (batch, side).
// embedding -> 3x (QKV, WMMA attention, softmax, att@v, residual relu) -> DNN
// ---------------------------------------------------------------------------
__global__ __launch_bounds__(256)
void fused_embed_att_dnn(const float* __restrict__ x,
                         const float* __restrict__ emb_w, const float* __restrict__ emb_b,
                         const float* __restrict__ Wq, const float* __restrict__ Wk,
                         const float* __restrict__ Wv,
                         const float* __restrict__ w1, const float* __restrict__ b1,
                         const float* __restrict__ w2, const float* __restrict__ b2,
                         const float* __restrict__ w3, const float* __restrict__ b3,
                         float* __restrict__ out32)
{
  __shared__ float e_s[MIDD][EDIM];
  __shared__ float q_s[MIDD][EDIM];
  __shared__ float k_s[MIDD][EDIM];
  __shared__ float v_s[MIDD][EDIM];
  __shared__ float o_s[MIDD][EDIM];
  __shared__ float w_s[3][64];
  __shared__ float red[64][4];
  __shared__ float h1s[64];
  __shared__ float h2s[32];

  const int tid   = threadIdx.x;
  const int b     = blockIdx.x >> 1;
  const int side  = blockIdx.x & 1;
  const int lane  = tid & 31;
  const int wave  = tid >> 5;
  const int lrow  = lane & 15;
  const int lhalf = lane >> 4;

  if (tid < 64) {
    w_s[0][tid] = Wq[tid];
    w_s[1][tid] = Wk[tid];
    w_s[2][tid] = Wv[tid];
  }
  // embedding: e[m][j] = x[b, side*128+m] * emb_w[m][j] + emb_b[m][j]
  const float* xrow = x + (size_t)b * FDIM + side * MIDD;
  for (int t = tid; t < MIDD * EDIM; t += 256) {
    int m = t >> 3;
    e_s[m][t & 7] = xrow[m] * emb_w[t] + emb_b[t];
  }
  __syncthreads();

  for (int iter = 0; iter < 3; ++iter) {
    // ---- QKV projections (8x8 weights, LDS resident) ----
    for (int t = tid; t < 3 * MIDD * EDIM; t += 256) {
      int sel = t >> 10;
      int rem = t & 1023;
      int m = rem >> 3, j = rem & 7;
      float s = 0.f;
#pragma unroll
      for (int i = 0; i < EDIM; ++i) s += e_s[m][i] * w_s[sel][i * EDIM + j];
      if (sel == 0)      q_s[m][j] = s;
      else if (sel == 1) k_s[m][j] = s;
      else               v_s[m][j] = s;
    }
    __syncthreads();

    // ---- attention: 16 tasks (head, qtile) over 8 waves x 2 reps ----
#pragma unroll
    for (int rep = 0; rep < 2; ++rep) {
      const int task = wave + 8 * rep;
      const int h    = task >> 3;
      const int qt   = task & 7;
      const int hb   = h * HDD;

      v8f acc[8];
#if __has_builtin(__builtin_amdgcn_wmma_f32_16x16x4_f32)
      // f32 16x16x4: A = q tile (16x4), B = k^T tile (4x16). Exact fp32 path.
      v2f a;
      a.x = q_s[qt * 16 + lrow][hb + 2 * lhalf];
      a.y = q_s[qt * 16 + lrow][hb + 2 * lhalf + 1];
#pragma unroll
      for (int kt = 0; kt < 8; ++kt) {
        v2f bm;
        bm.x = k_s[kt * 16 + lrow][hb + 2 * lhalf];
        bm.y = k_s[kt * 16 + lrow][hb + 2 * lhalf + 1];
        v8f cz = {};
        acc[kt] = __builtin_amdgcn_wmma_f32_16x16x4_f32(
            false, a, false, bm, (short)0, cz, false, false);
      }
#else
      // fallback: f16 16x16x32 with K zero-padded beyond HD=4
      v16h a16 = {};
      if (lhalf == 0) {
        a16[0] = (_Float16)q_s[qt * 16 + lrow][hb + 0];
        a16[1] = (_Float16)q_s[qt * 16 + lrow][hb + 1];
        a16[2] = (_Float16)q_s[qt * 16 + lrow][hb + 2];
        a16[3] = (_Float16)q_s[qt * 16 + lrow][hb + 3];
      }
#pragma unroll
      for (int kt = 0; kt < 8; ++kt) {
        v16h b16 = {};
        if (lhalf == 0) {
          b16[0] = (_Float16)k_s[kt * 16 + lrow][hb + 0];
          b16[1] = (_Float16)k_s[kt * 16 + lrow][hb + 1];
          b16[2] = (_Float16)k_s[kt * 16 + lrow][hb + 2];
          b16[3] = (_Float16)k_s[kt * 16 + lrow][hb + 3];
        }
        v8f cz = {};
        acc[kt] = __builtin_amdgcn_wmma_f32_16x16x32_f16(
            false, a16, false, b16, (short)0, cz, false, false);
      }
#endif
      // scale (1/sqrt(4)=0.5) + row softmax.
      // D layout: VGPR i, lane L -> (m = i + 8*(L>>4), n = kt*16 + (L&15))
      float rmax[8], rsum[8];
#pragma unroll
      for (int i = 0; i < 8; ++i) rmax[i] = -1e30f;
#pragma unroll
      for (int kt = 0; kt < 8; ++kt)
#pragma unroll
        for (int i = 0; i < 8; ++i) {
          acc[kt][i] *= 0.5f;
          rmax[i] = fmaxf(rmax[i], acc[kt][i]);
        }
#pragma unroll
      for (int i = 0; i < 8; ++i) { rmax[i] = grp16_max(rmax[i]); rsum[i] = 0.f; }
#pragma unroll
      for (int kt = 0; kt < 8; ++kt)
#pragma unroll
        for (int i = 0; i < 8; ++i) {
          acc[kt][i] = __expf(acc[kt][i] - rmax[i]);
          rsum[i] += acc[kt][i];
        }
#pragma unroll
      for (int i = 0; i < 8; ++i) rsum[i] = grp16_sum(rsum[i]);

      // o = p @ v   (K=128 via per-lane FMA + 16-lane reduce; only HD=4 cols)
      float accO[8][HDD];
#pragma unroll
      for (int i = 0; i < 8; ++i)
#pragma unroll
        for (int d = 0; d < HDD; ++d) accO[i][d] = 0.f;
#pragma unroll
      for (int kt = 0; kt < 8; ++kt) {
        int n = kt * 16 + lrow;
#pragma unroll
        for (int d = 0; d < HDD; ++d) {
          float vv = v_s[n][hb + d];
#pragma unroll
          for (int i = 0; i < 8; ++i) accO[i][d] += acc[kt][i] * vv;
        }
      }
#pragma unroll
      for (int i = 0; i < 8; ++i) {
#pragma unroll
        for (int d = 0; d < HDD; ++d) accO[i][d] = grp16_sum(accO[i][d]);
        float inv = 1.f / rsum[i];
        int m = qt * 16 + i + 8 * lhalf;
#pragma unroll
        for (int d = 0; d < HDD; ++d)
          if (lrow == d) o_s[m][hb + d] = accO[i][d] * inv;
      }
    }
    __syncthreads();
    // residual + relu
    for (int t = tid; t < MIDD * EDIM; t += 256) {
      int m = t >> 3, j = t & 7;
      float vv = e_s[m][j] + o_s[m][j];
      e_s[m][j] = vv > 0.f ? vv : 0.f;
    }
    __syncthreads();
  }

  // ---- DNN: 1024 -> 64 -> 32 -> 16 ----
  const float* e_flat = &e_s[0][0];
  {
    int o = tid & 63, c = tid >> 6;
    float s = 0.f;
    int f0 = c * 256;
    for (int f = 0; f < 256; ++f) {
      int ff = f0 + f;
      s += e_flat[ff] * w1[(size_t)ff * 64 + o];
    }
    red[o][c] = s;
  }
  __syncthreads();
  if (tid < 64) {
    float s = b1[tid] + red[tid][0] + red[tid][1] + red[tid][2] + red[tid][3];
    h1s[tid] = s > 0.f ? s : 0.f;
  }
  __syncthreads();
  if (tid < 32) {
    float s = b2[tid];
    for (int i = 0; i < 64; ++i) s += h1s[i] * w2[i * 32 + tid];
    h2s[tid] = s > 0.f ? s : 0.f;
  }
  __syncthreads();
  if (tid < 16) {
    float s = b3[tid];
    for (int i = 0; i < 32; ++i) s += h2s[i] * w3[i * 16 + tid];
    out32[(size_t)b * 32 + side * 16 + tid] = s > 0.f ? s : 0.f;
  }
}

// ---------------------------------------------------------------------------
// Kernel 2: co-interaction tail. One wave (32 lanes) per batch element.
// ---------------------------------------------------------------------------
__global__ __launch_bounds__(256)
void co_interact(const float* __restrict__ out32,
                 const float* __restrict__ co_w, const float* __restrict__ co_b,
                 const float* __restrict__ M,
                 const float* __restrict__ co_dw, const float* __restrict__ co_db,
                 float* __restrict__ out)
{
  __shared__ float G_s[8][16][16];
  __shared__ float M_s[64];
  __shared__ float o_sc[8][32];

  const int tid  = threadIdx.x;
  const int lane = tid & 31;
  const int wave = tid >> 5;
  const int b    = blockIdx.x * 8 + wave;

  if (tid < 64) M_s[tid] = M[tid];
  __syncthreads();

  float ov = out32[(size_t)b * 32 + lane];
  float er[EDIM];
#pragma unroll
  for (int j = 0; j < EDIM; ++j)
    er[j] = ov * co_w[lane * EDIM + j] + co_b[lane * EDIM + j];

  // lanes 0-15 hold gl rows; compute gl @ M
  float gm[EDIM];
#pragma unroll
  for (int e = 0; e < EDIM; ++e) {
    float s = 0.f;
#pragma unroll
    for (int d = 0; d < EDIM; ++d) s += er[d] * M_s[d * EDIM + e];
    gm[e] = s;
  }
  // G[l][r] = tanh((gl@M) . gr) ; gr rows live in lanes 16..31
  for (int r = 0; r < 16; ++r) {
    float g = 0.f;
#pragma unroll
    for (int j = 0; j < EDIM; ++j) {
      float grj = __shfl(er[j], 16 + r, 32);
      g += gm[j] * grj;
    }
    if (lane < 16) G_s[wave][lane][r] = tanhf(g);
  }
  __syncthreads();

  float lr = 0.f, rl = 0.f;
  if (lane < 16) {
    float cs = 0.f, rs = 0.f;
    for (int l = 0; l < 16; ++l) cs += G_s[wave][l][lane];   // sum over l -> lr
    for (int r = 0; r < 16; ++r) rs += G_s[wave][lane][r];   // sum over r -> rl
    float mcs = grp16_max(cs), mrs = grp16_max(rs);
    cs = __expf(cs - mcs);
    rs = __expf(rs - mrs);
    float scs = grp16_sum(cs), srs = grp16_sum(rs);
    lr = cs / scs;
    rl = rs / srs;
  }
  float slr = __shfl(lr, lane & 15, 32);
  float srl = __shfl(rl, lane & 15, 32);
  float scale = (lane < 16) ? slr : srl;
  o_sc[wave][lane] = ov * scale;
  __syncthreads();

  if (lane < 16) {
    float s = co_db[lane];
    for (int i = 0; i < 32; ++i) s += o_sc[wave][i] * co_dw[i * 16 + lane];
    out[(size_t)b * 16 + lane] = s > 0.f ? s : 0.f;
  }
}

extern "C" void kernel_launch(void* const* d_in, const int* in_sizes, int n_in,
                              void* d_out, int out_size, void* d_ws, size_t ws_size,
                              hipStream_t stream) {
  const float* x     = (const float*)d_in[0];
  const float* emb_w = (const float*)d_in[1];
  const float* emb_b = (const float*)d_in[2];
  const float* Wq    = (const float*)d_in[3];
  const float* Wk    = (const float*)d_in[4];
  const float* Wv    = (const float*)d_in[5];
  const float* w1    = (const float*)d_in[6];
  const float* b1    = (const float*)d_in[7];
  const float* w2    = (const float*)d_in[8];
  const float* b2    = (const float*)d_in[9];
  const float* w3    = (const float*)d_in[10];
  const float* b3    = (const float*)d_in[11];
  const float* co_w  = (const float*)d_in[12];
  const float* co_b  = (const float*)d_in[13];
  const float* M     = (const float*)d_in[14];
  const float* co_dw = (const float*)d_in[15];
  const float* co_db = (const float*)d_in[16];
  float* out32 = (float*)d_ws;                    // B*32 floats scratch
  float* out   = (float*)d_out;                   // B*16 floats

  fused_embed_att_dnn<<<dim3(2 * B_TOT), 256, 0, stream>>>(
      x, emb_w, emb_b, Wq, Wk, Wv, w1, b1, w2, b2, w3, b3, out32);
  co_interact<<<dim3(B_TOT / 8), 256, 0, stream>>>(
      out32, co_w, co_b, M, co_dw, co_db, out);
}
